// SpectralInverse_20005957665293
// MI455X (gfx1250) — matrix operand order repaired and checked by reading, not compiled, not run on hardware
//
#include <hip/hip_runtime.h>

typedef __attribute__((ext_vector_type(16))) _Float16 v16h;
typedef __attribute__((ext_vector_type(8)))  _Float16 v8h;
typedef __attribute__((ext_vector_type(8)))  float    v8f;

#define F_DIM   128
#define CIN     64
#define COUT    8
#define NBATCH  4
#define NGRID   110592          // 48*48*48
#define PTS_BLK 128             // points per workgroup (8 M-tiles of 16)
#define CSTR    136             // LDS stride (halves) for cos tile, padded vs 128
#define HSTR    136             // LDS stride (halves) for hT, padded vs 128
#define USTR    68              // LDS stride (floats) for U tile, padded vs 64
#define INV_TWO_PI 0.15915494309189535f

__global__ __launch_bounds__(256) void spectral_inverse_fused(
    const float* __restrict__ hg,   // (B,F,Cin)
    const float* __restrict__ yg,   // (B,48,48,48,3)
    const float* __restrict__ ywg,  // (F,3)
    const float* __restrict__ ybg,  // (F)
    const float* __restrict__ uwg,  // (Cout,Cin)
    const float* __restrict__ ubg,  // (Cout)
    float* __restrict__ out)        // (B,48,48,48,Cout)
{
    __shared__ _Float16 Clds[PTS_BLK * CSTR];  // cos tile, f16, A-operand source
    __shared__ _Float16 hT[CIN * HSTR];        // h[b] transposed [c][f], f16, B-operand source
    __shared__ float    Us[PTS_BLK * USTR];    // sin(U/N) tile, f32
    __shared__ float    ywl[F_DIM * 3];
    __shared__ float    ybl[F_DIM];
    __shared__ float    uwl[COUT * CIN];
    __shared__ float    ubl[COUT];
    __shared__ float    pts[PTS_BLK * 3];

    const int t     = threadIdx.x;
    const int b     = blockIdx.y;
    const int gbase = blockIdx.x * PTS_BLK;    // linear grid-point base within batch

    // ---------------- Phase 0: stage operands into LDS ----------------
    for (int idx = t; idx < F_DIM * CIN; idx += 256) {
        int f = idx >> 6, c = idx & 63;                       // h[b][f][c]
        hT[c * HSTR + f] = (_Float16)hg[b * F_DIM * CIN + idx];
    }
    for (int idx = t; idx < F_DIM * 3; idx += 256) ywl[idx] = ywg[idx];
    if (t < F_DIM) ybl[t] = ybg[t];
    for (int idx = t; idx < COUT * CIN; idx += 256) uwl[idx] = uwg[idx];
    if (t < COUT) ubl[t] = ubg[t];
    for (int idx = t; idx < PTS_BLK * 3; idx += 256)
        pts[idx] = yg[(size_t)(b * NGRID + gbase) * 3 + idx];
    __syncthreads();

    // ------------- Phase 1: C = cos(2*pi*(y.y_w + y_b)), f16 tile -------------
    // 128 points x 128 freqs = 16384 values; 64 per thread (one point-half each)
    {
        int p   = t >> 1;             // 0..127 point
        int fb0 = (t & 1) << 6;       // 0 or 64
        float x0 = pts[p * 3 + 0], x1 = pts[p * 3 + 1], x2 = pts[p * 3 + 2];
#pragma unroll
        for (int ch = 0; ch < 4; ++ch) {
            int fb = fb0 + ch * 16;
            alignas(32) _Float16 cv[16];
#pragma unroll
            for (int i = 0; i < 16; ++i) {
                int f = fb + i;
                float Yv = __builtin_fmaf(x0, ywl[f * 3 + 0],
                           __builtin_fmaf(x1, ywl[f * 3 + 1],
                           __builtin_fmaf(x2, ywl[f * 3 + 2], ybl[f])));
                // v_cos_f32 computes cos(2*pi*src) -> exactly cos(2*pi*Y), |Y|<256 OK
                cv[i] = (_Float16)__builtin_amdgcn_cosf(Yv);
            }
            *(v8h*)&Clds[p * CSTR + fb]     = *(const v8h*)&cv[0];
            *(v8h*)&Clds[p * CSTR + fb + 8] = *(const v8h*)&cv[8];
        }
    }
    __syncthreads();

    // ------------- Phase 2: U = C x h[b] via WMMA, then sin activation -------------
    // 8 waves: wave = (m_group 0..1) x (n_tile 0..3); each wave does 4 M-tiles
    // of 16 points x 16 Cin channels, K=128 in 4 WMMA steps -> 16 WMMAs/wave.
    {
        const int wv     = t >> 5;           // wave id 0..7
        const int lane   = t & 31;
        const int n_tile = wv & 3;           // 16 Cin channels each
        const int m_grp  = wv >> 2;          // 64 points each

        // B fragments (32x16 layout): lanes<16 -> K {0..7,16..23}, lanes>=16 -> K {8..15,24..31}.
        // Load once, reuse across all 4 M-tiles.
        const int colB = n_tile * 16 + (lane & 15);
        const int kB   = (lane < 16) ? 0 : 8;
        v16h bfrag[4];
#pragma unroll
        for (int kc = 0; kc < 4; ++kc) {
            v8h blo = *(const v8h*)&hT[colB * HSTR + kc * 32 + kB];
            v8h bhi = *(const v8h*)&hT[colB * HSTR + kc * 32 + kB + 16];
            bfrag[kc] = __builtin_shufflevector(blo, bhi,
                          0,1,2,3,4,5,6,7,8,9,10,11,12,13,14,15);
        }

        // A fragment addressing (16-bit 16x32 layout): lane<16 -> K {0..7,16..23}
        const int kA     = (lane < 16) ? 0 : 8;
        const int c      = n_tile * 16 + (lane & 15);
        const float sscale = INV_TWO_PI / (float)NGRID;

#pragma unroll
        for (int mt = 0; mt < 4; ++mt) {
            const int rowA = m_grp * 64 + mt * 16 + (lane & 15);
            v8f acc = {};
#pragma unroll
            for (int kc = 0; kc < 4; ++kc) {   // K = 128 in 4 steps of 32
                v8h alo = *(const v8h*)&Clds[rowA * CSTR + kc * 32 + kA];
                v8h ahi = *(const v8h*)&Clds[rowA * CSTR + kc * 32 + kA + 16];
                v16h a  = __builtin_shufflevector(alo, ahi,
                            0,1,2,3,4,5,6,7,8,9,10,11,12,13,14,15);
                acc = __builtin_amdgcn_wmma_f32_16x16x32_f16(
                          false, a, false, bfrag[kc], (short)0, acc, false, false);
            }
            // activation: sin(U / NGRID) == v_sin_f32(U / (NGRID * 2pi))
            const int rbase = m_grp * 64 + mt * 16 + ((lane < 16) ? 0 : 8);
#pragma unroll
            for (int j = 0; j < 8; ++j) {
                Us[(rbase + j) * USTR + c] = __builtin_amdgcn_sinf(acc[j] * sscale);
            }
        }
    }
    __syncthreads();

    // ------------- Phase 3: out = Us . u_w^T + u_b (Cout=8, f32 VALU) -------------
    {
        int pb = t >> 3;                      // 0..31
        int o  = t & 7;                       // 0..7
#pragma unroll
        for (int pp = 0; pp < 4; ++pp) {
            int p = pb + pp * 32;
            float s = ubl[o];
#pragma unroll 8
            for (int c = 0; c < CIN; ++c)
                s = __builtin_fmaf(Us[p * USTR + c], uwl[o * CIN + c], s);
            out[(size_t)(b * NGRID + gbase + p) * COUT + o] = s;
        }
    }
}

extern "C" void kernel_launch(void* const* d_in, const int* in_sizes, int n_in,
                              void* d_out, int out_size, void* d_ws, size_t ws_size,
                              hipStream_t stream) {
    const float* hg  = (const float*)d_in[0];  // h   (4,128,64)
    const float* yg  = (const float*)d_in[1];  // y   (4,48,48,48,3)
    const float* ywg = (const float*)d_in[2];  // y_w (128,3)
    const float* ybg = (const float*)d_in[3];  // y_b (128)
    const float* uwg = (const float*)d_in[4];  // u_w (8,64)
    const float* ubg = (const float*)d_in[5];  // u_b (8)
    float* out = (float*)d_out;                // (4,48,48,48,8)

    dim3 grid(NGRID / PTS_BLK, NBATCH);        // 864 x 4 blocks
    spectral_inverse_fused<<<grid, 256, 0, stream>>>(hg, yg, ywg, ybg, uwg, ubg, out);
}
